// Attention_53223234732453
// MI455X (gfx1250) — matrix operand order repaired and checked by reading, not compiled, not run on hardware
//
#include <hip/hip_runtime.h>

// ---------------------------------------------------------------------------
// CDNA5 (gfx1250) fused multi-head attention with relative-position bias.
// All matmuls use v_wmma_f32_16x16x32_bf16 (wave32 WMMA).
//   B=16, C=128, HEADS=4, d=32, H=W=32, N=1024.
// Identities used:
//   softmax(q·k*s + q·r*s) = softmax((q*s)·(k+r))      -> fold r into K
//   exp(x) = exp2(x*log2e)                             -> fold s*log2e into Q
// ---------------------------------------------------------------------------

typedef __attribute__((ext_vector_type(16))) __bf16 v16bf;
typedef __attribute__((ext_vector_type(8)))  float  v8f;

__device__ __forceinline__ v8f wmma_bf16(v16bf a, v16bf b, v8f c) {
  // D(16x16 f32) = A(16x32 bf16) * B(32x16 bf16) + C
  return __builtin_amdgcn_wmma_f32_16x16x32_bf16(
      /*neg_a=*/false, a, /*neg_b=*/false, b,
      /*c_mod=*/(short)0, c, /*reuse_a=*/false, /*reuse_b=*/false);
}

__device__ __forceinline__ float rowmax16(float v) {
  v = fmaxf(v, __shfl_xor(v, 1, 32));
  v = fmaxf(v, __shfl_xor(v, 2, 32));
  v = fmaxf(v, __shfl_xor(v, 4, 32));
  v = fmaxf(v, __shfl_xor(v, 8, 32));
  return v;
}
__device__ __forceinline__ float rowsum16(float v) {
  v += __shfl_xor(v, 1, 32);
  v += __shfl_xor(v, 2, 32);
  v += __shfl_xor(v, 4, 32);
  v += __shfl_xor(v, 8, 32);
  return v;
}

enum : int { kB = 16, kC = 128, kHeads = 4, kD = 32, kN = 1024 };

// ---------------------------------------------------------------------------
// Casts: W_qkv (384x128) f32->bf16, x (B,C,N) f32 -> x_t (B,N,C) bf16.
// ---------------------------------------------------------------------------
__global__ void cvt_w_kernel(const float* __restrict__ W, __bf16* __restrict__ Wt) {
  int i = blockIdx.x * 256 + threadIdx.x;
  if (i < 3 * kC * kC) Wt[i] = (__bf16)W[i];
}

__global__ void cvt_x_kernel(const float* __restrict__ X, __bf16* __restrict__ Xt) {
  int i = blockIdx.x * 256 + threadIdx.x;
  if (i < kB * kC * kN) {
    int b = i >> 17;            // / (128*1024)
    int c = (i >> 10) & 127;
    int n = i & 1023;
    Xt[((size_t)b * kN + n) * kC + c] = (__bf16)X[i];
  }
}

// ---------------------------------------------------------------------------
// QKV projection GEMM via WMMA.  qkv[b,o,n] = sum_c W[o,c] * x[b,c,n].
// One wave per 16x16 (o,n) tile, K=128 in 4 WMMA steps.
// Epilogue (wave-uniform branch; 16-row o-tile never straddles q/k/v):
//   Q <- q*scale*log2e  [bh][n][d]   (one b128 store per lane)
//   K <- k + (rw+rh)    [bh][n][d]   (one b128 store per lane)
//   V <- v              [bh][d][n]   (b16 scatter)
// ---------------------------------------------------------------------------
__global__ void qkv_gemm_kernel(const __bf16* __restrict__ Wt,
                                const __bf16* __restrict__ Xt,
                                const float* __restrict__ rw,
                                const float* __restrict__ rh,
                                __bf16* __restrict__ Q,
                                __bf16* __restrict__ K,
                                __bf16* __restrict__ V) {
  const int wave = threadIdx.x >> 5;
  const int lane = threadIdx.x & 31;
  const int g    = lane >> 4;     // half-wave group
  const int lr   = lane & 15;

  const int job = blockIdx.x * 4 + wave;        // 24576 jobs total
  const int b   = job / 1536;                   // 24 o-tiles * 64 n-tiles
  const int rem = job % 1536;
  const int o0  = (rem >> 6) * 16;              // 0..368
  const int n0  = (rem & 63) * 16;              // 0..1008

  const __bf16* wrow = Wt + (size_t)(o0 + lr) * kC;
  const __bf16* xrow = Xt + ((size_t)b * kN + n0 + lr) * kC + g * 16;

  v8f acc = {};
#pragma unroll
  for (int c0 = 0; c0 < kC; c0 += 32) {
    v16bf a, bm;
    // A (16x32): lanes 0-15 K=c0+0..15, lanes 16-31 K=c0+16..31
    __builtin_memcpy(&a, wrow + c0 + g * 16, 16);
    __builtin_memcpy((char*)&a + 16, wrow + c0 + g * 16 + 8, 16);
    // B (32x16): lane n=n0+lr holds 16 consecutive c at c0 + g*16
    __builtin_memcpy(&bm, xrow + c0, 32);
    acc = wmma_bf16(a, bm, acc);
  }

  // logits computed as exp2, so fold 1/sqrt(d) * log2(e) into q
  const float kQScale = 0.17677669529663687f * 1.4426950408889634f;
  const int n = n0 + lr;
  const int wimg = n & 31, himg = n >> 5;

  const int seg     = o0 >> 7;                 // 0=q, 1=k, 2=v (wave-uniform)
  const int h       = (o0 & 127) >> 5;
  const int ddstart = (o0 & 31) + g * 8;       // 8 consecutive dd per lane
  const size_t bh   = (size_t)b * kHeads + h;

  if (seg == 0) {
    __bf16 pk[8];
#pragma unroll
    for (int r = 0; r < 8; ++r) pk[r] = (__bf16)(acc[r] * kQScale);
    __builtin_memcpy(Q + (bh * kN + n) * kD + ddstart, pk, 16);
  } else if (seg == 1) {
    __bf16 pk[8];
#pragma unroll
    for (int r = 0; r < 8; ++r) {
      const int dd = ddstart + r;
      const float rv = rw[h * 1024 + dd * 32 + wimg] +
                       rh[h * 1024 + dd * 32 + himg];
      pk[r] = (__bf16)(acc[r] + rv);
    }
    __builtin_memcpy(K + (bh * kN + n) * kD + ddstart, pk, 16);
  } else {
#pragma unroll
    for (int r = 0; r < 8; ++r)
      V[(bh * kD + ddstart + r) * kN + n] = (__bf16)acc[r];
  }
}

// ---------------------------------------------------------------------------
// Flash attention: one wave owns a 16-query tile, streams 1024 keys in chunks
// of 64 (4 S-tiles per online-softmax update -> half the shuffle/rescale cost
// per key vs 32-wide chunks).  4 waves/block, 1024 blocks.
// ---------------------------------------------------------------------------
__global__ void flash_attn_kernel(const __bf16* __restrict__ Q,
                                  const __bf16* __restrict__ K,
                                  const __bf16* __restrict__ V,
                                  float* __restrict__ out) {
  __shared__ __align__(16) __bf16 stage[4][16 * 64];  // per-wave P transpose

  const int wave = threadIdx.x >> 5;
  const int lane = threadIdx.x & 31;
  const int g    = lane >> 4;
  const int lr   = lane & 15;

  const int bh    = blockIdx.x >> 4;   // 0..63  (b*4 + h)
  const int chunk = blockIdx.x & 15;
  const int b = bh >> 2, h = bh & 3;
  const int i0 = chunk * 64 + wave * 16;

  // Q fragment (A layout); scale*log2e already folded in.
  v16bf aq;
  {
    const __bf16* qrow = Q + ((size_t)bh * kN + i0 + lr) * kD;
    __builtin_memcpy(&aq, qrow + g * 16, 16);
    __builtin_memcpy((char*)&aq + 16, qrow + g * 16 + 8, 16);
  }

  float m[8], l[8];
  v8f oa = {}, ob = {};  // output accumulators for dd 0..15 / 16..31
#pragma unroll
  for (int r = 0; r < 8; ++r) { m[r] = -3.0e38f; l[r] = 0.0f; }

  const __bf16* Kbh = K + (size_t)bh * kN * kD;
  const __bf16* Vbh = V + (size_t)bh * kD * kN;
  __bf16* st = stage[wave];
  const v8f zero = {};

  for (int j0 = 0; j0 < kN; j0 += 64) {
    // --- logits: four 16-key tiles (base-2 units) ---
    v8f s[4];
#pragma unroll
    for (int t = 0; t < 4; ++t) {
      v16bf bk;
      __builtin_memcpy(&bk, Kbh + (size_t)(j0 + t * 16 + lr) * kD + g * 16, 32);
      s[t] = wmma_bf16(aq, bk, zero);
    }

    // --- online softmax, rows M = r + 8*g live in lanes of one half ---
#pragma unroll
    for (int r = 0; r < 8; ++r) {
      const float a0 = s[0][r], a1 = s[1][r], a2 = s[2][r], a3 = s[3][r];
      const float tm   = rowmax16(fmaxf(fmaxf(a0, a1), fmaxf(a2, a3)));
      const float mnew = fmaxf(m[r], tm);
      const float f  = __builtin_amdgcn_exp2f(m[r] - mnew);
      const float p0 = __builtin_amdgcn_exp2f(a0 - mnew);
      const float p1 = __builtin_amdgcn_exp2f(a1 - mnew);
      const float p2 = __builtin_amdgcn_exp2f(a2 - mnew);
      const float p3 = __builtin_amdgcn_exp2f(a3 - mnew);
      l[r] = l[r] * f + rowsum16((p0 + p1) + (p2 + p3));
      m[r] = mnew;
      oa[r] *= f;
      ob[r] *= f;
      const int row = r + g * 8;               // D layout -> row-major stage
      st[row * 64 + lr]      = (__bf16)p0;
      st[row * 64 + 16 + lr] = (__bf16)p1;
      st[row * 64 + 32 + lr] = (__bf16)p2;
      st[row * 64 + 48 + lr] = (__bf16)p3;
    }

    // D-layout -> A-layout transpose bounce through wave-private LDS.
    // Same-wave LDS ops are in-order on CDNA5; just stop compiler reordering.
    asm volatile("" ::: "memory");
    v16bf ap0, ap1;
    {
      const __bf16* prow = st + lr * 64;
      __builtin_memcpy(&ap0, prow + g * 16, 16);
      __builtin_memcpy((char*)&ap0 + 16, prow + g * 16 + 8, 16);
      __builtin_memcpy(&ap1, prow + 32 + g * 16, 16);
      __builtin_memcpy((char*)&ap1 + 16, prow + 32 + g * 16 + 8, 16);
    }
    asm volatile("" ::: "memory");

    // --- O += P(16x64) x V^T(64x16), two dd halves x two j sub-chunks ---
    v16bf bv;
    __builtin_memcpy(&bv, Vbh + (size_t)lr * kN + j0 + g * 16, 32);
    oa = wmma_bf16(ap0, bv, oa);
    __builtin_memcpy(&bv, Vbh + (size_t)lr * kN + j0 + 32 + g * 16, 32);
    oa = wmma_bf16(ap1, bv, oa);
    __builtin_memcpy(&bv, Vbh + (size_t)(16 + lr) * kN + j0 + g * 16, 32);
    ob = wmma_bf16(ap0, bv, ob);
    __builtin_memcpy(&bv, Vbh + (size_t)(16 + lr) * kN + j0 + 32 + g * 16, 32);
    ob = wmma_bf16(ap1, bv, ob);
  }

  // Normalize and write out[b, h*32+dd, i] (f32).  Per lane the 8 r-values
  // hit 8 consecutive i -> pack into one 32B store per dd-half.
  float fa[8], fb[8];
#pragma unroll
  for (int r = 0; r < 8; ++r) {
    const float inv = 1.0f / l[r];
    fa[r] = oa[r] * inv;
    fb[r] = ob[r] * inv;
  }
  const int ibase = i0 + g * 8;
  __builtin_memcpy(out + ((size_t)b * kC + h * kD + lr) * kN + ibase, fa, 32);
  __builtin_memcpy(out + ((size_t)b * kC + h * kD + 16 + lr) * kN + ibase, fb, 32);
}

// ---------------------------------------------------------------------------
// Launch
// ---------------------------------------------------------------------------
extern "C" void kernel_launch(void* const* d_in, const int* in_sizes, int n_in,
                              void* d_out, int out_size, void* d_ws, size_t ws_size,
                              hipStream_t stream) {
  const float* x  = (const float*)d_in[0];   // (16,128,32,32)
  const float* Wq = (const float*)d_in[1];   // (384,128)
  const float* rw = (const float*)d_in[2];   // (1,4,32,1,32)
  const float* rh = (const float*)d_in[3];   // (1,4,32,32,1)
  float* out = (float*)d_out;                // (16,128,32,32) f32

  char* ws = (char*)d_ws;
  __bf16* Xt = (__bf16*)(ws);                       //  4 MB: x^T  [b][n][c]
  __bf16* Wt = (__bf16*)(ws + 4194304);             // 96 KB: W bf16
  __bf16* Qw = (__bf16*)(ws + 4292608);             //  4 MB: q*scale [bh][n][d]
  __bf16* Kw = (__bf16*)(ws + 8486912);             //  4 MB: k+r     [bh][n][d]
  __bf16* Vw = (__bf16*)(ws + 12681216);            //  4 MB: v       [bh][d][n]

  cvt_x_kernel<<<8192, 256, 0, stream>>>(x, Xt);        // 2M elems
  cvt_w_kernel<<<192, 256, 0, stream>>>(Wq, Wt);        // 48K elems
  // 16 batches * 24 o-tiles * 64 n-tiles = 24576 wave-jobs, 4 waves/block
  qkv_gemm_kernel<<<6144, 128, 0, stream>>>(Wt, Xt, rw, rh, Qw, Kw, Vw);
  // 64 (b,h) * 16 query-chunks, 4 waves/block, 16 queries/wave
  flash_attn_kernel<<<1024, 128, 0, stream>>>(Qw, Kw, Vw, out);
}
